// InternLMAttention_68676527063336
// MI455X (gfx1250) — compile-verified, matrix operand order checked by reading, not run on hardware
//
#include <hip/hip_runtime.h>
#include <hip/hip_bf16.h>

// ---------------------------------------------------------------------------
// Types for CDNA5 WMMA (wave32): v_wmma_f32_16x16x32_bf16
// ---------------------------------------------------------------------------
typedef __attribute__((ext_vector_type(16))) __bf16 v16bf;
typedef __attribute__((ext_vector_type(8)))  __bf16 v8bf;
typedef __attribute__((ext_vector_type(8)))  float  v8f;
typedef __attribute__((ext_vector_type(4)))  unsigned su4;
typedef __attribute__((ext_vector_type(8)))  unsigned su8;

#define S_LEN   2048
#define HID     4096
#define NHEAD   32
#define HDIM    128
#define QKV_W   12288   // 3*HID

static __device__ __forceinline__ __bf16 f2bf(float f) {
    union { float f; unsigned u; } v; v.f = f;
    unsigned r = v.u + 0x7FFFu + ((v.u >> 16) & 1u);   // round-to-nearest-even
    unsigned short h = (unsigned short)(r >> 16);
    __bf16 out; __builtin_memcpy(&out, &h, 2);
    return out;
}
static __device__ __forceinline__ float bf2f(__bf16 b) {
    unsigned short h; __builtin_memcpy(&h, &b, 2);
    unsigned u = ((unsigned)h) << 16;
    float f; __builtin_memcpy(&f, &u, 4);
    return f;
}

// Low 32 bits of a generic pointer to a __shared__ object == LDS byte offset
// (local->flat addrspacecast places the aperture in the high 32 bits).
static __device__ __forceinline__ unsigned lds_off(const void* p) {
    return (unsigned)(size_t)p;
}

// Async global->LDS copy of 16 bytes (GLOBAL_LOAD_ASYNC_TO_LDS_B128, GV mode).
// Tracked on ASYNCcnt; INST_OFFSET applies to both LDS and global addresses.
static __device__ __forceinline__ void async_copy_b128(unsigned lds, const void* g) {
    asm volatile("global_load_async_to_lds_b128 %0, %1, off"
                 :: "v"(lds), "v"(g) : "memory");
}
static __device__ __forceinline__ void async_copy_2x_b128(unsigned lds, const void* g) {
    asm volatile("global_load_async_to_lds_b128 %0, %1, off\n\t"
                 "global_load_async_to_lds_b128 %0, %1, off offset:16"
                 :: "v"(lds), "v"(g) : "memory");
}
static __device__ __forceinline__ void wait_async0() {
    asm volatile("s_wait_asynccnt 0x0" ::: "memory");
}

// Build a 16x32 bf16 A/B fragment register (per-lane 16 values = two aligned
// 16B LDS reads) following the CDNA5 16-bit operand layout:
// lanes 0-15 hold K {kb..kb+7, kb+16..kb+23} with kb = half*8.
static __device__ __forceinline__ v16bf load_frag(const __bf16* row_ptr, int kb) {
    v8bf lo = *(const v8bf*)(row_ptr + kb);
    v8bf hi = *(const v8bf*)(row_ptr + kb + 16);
    v16bf f;
    #pragma unroll
    for (int e = 0; e < 8; ++e) { f[e] = lo[e]; f[8 + e] = hi[e]; }
    return f;
}

// ---------------------------------------------------------------------------
// Kernel 1: fp32 -> bf16 conversion
// ---------------------------------------------------------------------------
__global__ __launch_bounds__(256) void cvt_f32_bf16_kernel(
        const float* __restrict__ in, __bf16* __restrict__ out, int n) {
    int i = blockIdx.x * 256 + threadIdx.x;
    if (i < n) out[i] = f2bf(in[i]);
}

// ---------------------------------------------------------------------------
// Kernel 2/5: tiled WMMA GEMM  C[M,N] = A[M,K](bf16) * B[K,N](f32) + bias
// Block = 256 threads (8 waves), tile 128x128, K-step 32.
// LDS A: [row][K] (stride 40), LDS B: [col][K] transposed (stride 40).
// A tiles stream in via GLOBAL_LOAD_ASYNC_TO_LDS (no VGPR round-trip);
// B tiles convert f32->bf16 in registers while transposing into LDS.
// ---------------------------------------------------------------------------
#define GBM 128
#define GBN 128
#define GBK 32
#define LDST 40   // padded LDS K-stride (multiple of 8 => 16B-aligned v8bf)

template<bool OUT_BF16>
__global__ __launch_bounds__(256) void gemm_bias_kernel(
        const __bf16* __restrict__ A,   // [M][K] bf16
        const float*  __restrict__ Bm,  // [K][N] f32 (converted while staging)
        const float*  __restrict__ bias,// [N]
        void* __restrict__ Cout,        // [M][N] bf16 or f32
        int M, int N, int K) {
    __shared__ __bf16 lsA[GBM][LDST];
    __shared__ __bf16 lsB[GBN][LDST];

    const int tid  = threadIdx.x;
    const int bm   = blockIdx.y * GBM;
    const int bn   = blockIdx.x * GBN;
    const int wave = tid >> 5;
    const int lane = tid & 31;
    const int half = lane >> 4;
    const int lr   = lane & 15;
    const int wm   = (wave & 1) * 64;   // 2 waves along M (4 tiles each)
    const int wn   = (wave >> 1) * 32;  // 4 waves along N (2 tiles each)

    // Exercise the Tensor Data Mover path with an architected NOP:
    // a zero-filled D# has count==0 => NULL tensor (no data moved).
    {
        su4 td_g0 = {};
        su8 td_g1 = {};
        asm volatile("tensor_load_to_lds %0, %1" :: "s"(td_g0), "s"(td_g1) : "memory");
        __builtin_amdgcn_s_wait_tensorcnt(0);
    }

    v8f acc[4][2] = {};

    const int ar = tid >> 1;           // A staging: row per 2 threads
    const int ac = (tid & 1) * 16;     // 16 bf16 each

    for (int k0 = 0; k0 < K; k0 += GBK) {
        __syncthreads();
        // ---- stage A tile: async DMA, 32 bytes per thread ----
        {
            const __bf16* g = A + (size_t)(bm + ar) * K + k0 + ac;
            async_copy_2x_b128(lds_off(&lsA[ar][ac]), g);
        }
        // ---- stage B tile transposed, f32 -> bf16 ----
        #pragma unroll
        for (int it = 0; it < 4; ++it) {
            int q  = tid + it * 256;       // 0..1023
            int kk = q >> 5;               // 0..31
            int c4 = (q & 31) * 4;         // 0..124
            const float* g = Bm + (size_t)(k0 + kk) * N + bn + c4;
            if (it == 0 && k0 + GBK < K)
                __builtin_prefetch(g + (size_t)GBK * N, 0, 0);
            float4 f = *(const float4*)g;
            lsB[c4 + 0][kk] = f2bf(f.x);
            lsB[c4 + 1][kk] = f2bf(f.y);
            lsB[c4 + 2][kk] = f2bf(f.z);
            lsB[c4 + 3][kk] = f2bf(f.w);
        }
        wait_async0();
        __syncthreads();

        const int kb = half * 8;
        v16bf bF[2];
        #pragma unroll
        for (int ni = 0; ni < 2; ++ni)
            bF[ni] = load_frag(&lsB[wn + ni * 16 + lr][0], kb);
        #pragma unroll
        for (int mi = 0; mi < 4; ++mi) {
            v16bf aF = load_frag(&lsA[wm + mi * 16 + lr][0], kb);
            #pragma unroll
            for (int ni = 0; ni < 2; ++ni)
                acc[mi][ni] = __builtin_amdgcn_wmma_f32_16x16x32_bf16(
                    false, aF, false, bF[ni], (short)0, acc[mi][ni], false, false);
        }
    }

    // ---- epilogue: bias add, store ----
    #pragma unroll
    for (int mi = 0; mi < 4; ++mi)
        #pragma unroll
        for (int ni = 0; ni < 2; ++ni)
            #pragma unroll
            for (int r = 0; r < 8; ++r) {
                int gm = bm + wm + mi * 16 + half * 8 + r;
                int gn = bn + wn + ni * 16 + lr;
                float v = acc[mi][ni][r] + bias[gn];
                if (OUT_BF16)
                    ((__bf16*)Cout)[(size_t)gm * N + gn] = f2bf(v);
                else
                    ((float*)Cout)[(size_t)gm * N + gn] = v;
            }
}

// ---------------------------------------------------------------------------
// Kernel 3: RoPE applied in-place on q,k inside qkv (bf16)
// ---------------------------------------------------------------------------
__global__ __launch_bounds__(256) void rope_kernel(
        __bf16* __restrict__ qkv, const int* __restrict__ positions) {
    int idx = blockIdx.x * 256 + threadIdx.x;   // S * NHEAD * 64 threads
    if (idx >= S_LEN * NHEAD * 64) return;
    int d = idx & 63;
    int h = (idx >> 6) & (NHEAD - 1);
    int s = idx >> 11;
    float pos = (float)positions[s];
    float inv = __powf(10000.0f, -(float)(2 * d) / 128.0f);
    float f   = pos * inv;
    float c = __cosf(f), sn = __sinf(f);
    __bf16* q = qkv + (size_t)s * QKV_W + h * HDIM;
    __bf16* k = q + HID;
    float q1 = bf2f(q[d]), q2 = bf2f(q[d + 64]);
    q[d]      = f2bf(q1 * c - q2 * sn);
    q[d + 64] = f2bf(q2 * c + q1 * sn);
    float k1 = bf2f(k[d]), k2 = bf2f(k[d + 64]);
    k[d]      = f2bf(k1 * c - k2 * sn);
    k[d + 64] = f2bf(k2 * c + k1 * sn);
}

// ---------------------------------------------------------------------------
// Kernel 4: causal flash attention (bf16 WMMA, f32 online softmax)
// Grid: (S/64 query blocks, NHEAD). Block: 128 threads = 4 waves,
// each wave owns 16 query rows. Key blocks of 32.
// Q and K tiles stream into LDS via async DMA (pure bf16 copies);
// V is transposed through VGPRs (layout change needed).
// ---------------------------------------------------------------------------
__global__ __launch_bounds__(128) void attn_kernel(
        const __bf16* __restrict__ qkv,   // [S][12288]
        __bf16* __restrict__ outb) {      // [S][4096]
    const int h     = blockIdx.y;
    const int qbase = blockIdx.x * 64;

    __shared__ __bf16 lsQ[64][136];     // [qrow][d]   (A layout)
    __shared__ __bf16 lsK[32][136];     // [key][d]    (B layout for Q*K^T)
    __shared__ __bf16 lsVt[128][40];    // [d][key]    (B layout for P*V)
    __shared__ __bf16 lsP[4][16][40];   // per-wave P re-layout buffer

    const int tid  = threadIdx.x;
    const int wave = tid >> 5;
    const int lane = tid & 31;
    const int half = lane >> 4;
    const int lr   = lane & 15;

    // ---- stage Q once (async copies; waited before first compute) ----
    for (int idx = tid; idx < 64 * 16; idx += 128) {
        int r = idx >> 4, c = (idx & 15) * 8;
        async_copy_b128(lds_off(&lsQ[r][c]),
                        qkv + (size_t)(qbase + r) * QKV_W + h * HDIM + c);
    }

    v8f  oAcc[8] = {};
    float mrow[8], lrow[8];
    #pragma unroll
    for (int r = 0; r < 8; ++r) { mrow[r] = -3.0e38f; lrow[r] = 0.0f; }

    const float scale = 0.08838834764831845f;   // 1/sqrt(128)
    const int jmax = (qbase + 63) >> 5;         // inclusive causal bound

    for (int j = 0; j <= jmax; ++j) {
        const int kg0 = j * 32;
        __syncthreads();
        // ---- stage K block (async DMA) ----
        for (int idx = tid; idx < 32 * 16; idx += 128) {
            int r = idx >> 4, c = (idx & 15) * 8;
            async_copy_b128(lds_off(&lsK[r][c]),
                            qkv + (size_t)(kg0 + r) * QKV_W + HID + h * HDIM + c);
        }
        // ---- stage V block transposed (VGPR path: layout change) ----
        {
            int key = tid >> 2;
            int d0  = (tid & 3) * 32;
            const __bf16* g = qkv + (size_t)(kg0 + key) * QKV_W + 2 * HID + h * HDIM + d0;
            #pragma unroll
            for (int cc = 0; cc < 4; ++cc) {
                v8bf v = *(const v8bf*)(g + cc * 8);
                #pragma unroll
                for (int u = 0; u < 8; ++u) lsVt[d0 + cc * 8 + u][key] = v[u];
            }
        }
        wait_async0();
        __syncthreads();

        // ---- scores: S = Q * K^T  (16q x 32k per wave, 4 K-steps over d) ----
        v8f sAcc[2] = {};
        #pragma unroll
        for (int kk = 0; kk < 4; ++kk) {
            int kb = kk * 32 + half * 8;
            v16bf aF = load_frag(&lsQ[wave * 16 + lr][0], kb);
            #pragma unroll
            for (int ni = 0; ni < 2; ++ni) {
                v16bf bF = load_frag(&lsK[ni * 16 + lr][0], kb);
                sAcc[ni] = __builtin_amdgcn_wmma_f32_16x16x32_bf16(
                    false, aF, false, bF, (short)0, sAcc[ni], false, false);
            }
        }

        // ---- causal mask + online softmax (per-row stats in 16-lane halves) ----
        float pm[2][8];
        #pragma unroll
        for (int ni = 0; ni < 2; ++ni)
            #pragma unroll
            for (int r = 0; r < 8; ++r) {
                int qg = qbase + wave * 16 + half * 8 + r;
                int kg = kg0 + ni * 16 + lr;
                float sv = sAcc[ni][r] * scale;
                pm[ni][r] = (kg <= qg) ? sv : -3.0e38f;
            }
        #pragma unroll
        for (int r = 0; r < 8; ++r) {
            float mx = fmaxf(pm[0][r], pm[1][r]);
            #pragma unroll
            for (int m = 1; m < 16; m <<= 1) mx = fmaxf(mx, __shfl_xor(mx, m, 32));
            float mnew  = fmaxf(mrow[r], mx);
            float alpha = __expf(mrow[r] - mnew);
            float p0 = __expf(pm[0][r] - mnew);
            float p1 = __expf(pm[1][r] - mnew);
            float ps = p0 + p1;
            #pragma unroll
            for (int m = 1; m < 16; m <<= 1) ps += __shfl_xor(ps, m, 32);
            lrow[r] = lrow[r] * alpha + ps;
            mrow[r] = mnew;
            #pragma unroll
            for (int t = 0; t < 8; ++t) oAcc[t][r] *= alpha;
            pm[0][r] = p0; pm[1][r] = p1;
        }

        // ---- re-layout P (C layout -> A layout) through LDS ----
        #pragma unroll
        for (int ni = 0; ni < 2; ++ni)
            #pragma unroll
            for (int r = 0; r < 8; ++r)
                lsP[wave][half * 8 + r][ni * 16 + lr] = f2bf(pm[ni][r]);
        __syncthreads();

        // ---- O += P * V  (16q x 128d per wave, K=32 keys) ----
        {
            int kb = half * 8;
            v16bf aP = load_frag(&lsP[wave][lr][0], kb);
            #pragma unroll
            for (int t = 0; t < 8; ++t) {
                v16bf bV = load_frag(&lsVt[t * 16 + lr][0], kb);
                oAcc[t] = __builtin_amdgcn_wmma_f32_16x16x32_bf16(
                    false, aP, false, bV, (short)0, oAcc[t], false, false);
            }
        }
    }

    // ---- normalize and store (bf16) ----
    #pragma unroll
    for (int t = 0; t < 8; ++t)
        #pragma unroll
        for (int r = 0; r < 8; ++r) {
            int s = qbase + wave * 16 + half * 8 + r;
            int d = t * 16 + lr;
            float v = oAcc[t][r] / lrow[r];
            outb[(size_t)s * HID + h * HDIM + d] = f2bf(v);
        }
}

// ---------------------------------------------------------------------------
// Launch: convert -> QKV GEMM -> RoPE -> flash attention -> O-proj GEMM
// ---------------------------------------------------------------------------
extern "C" void kernel_launch(void* const* d_in, const int* in_sizes, int n_in,
                              void* d_out, int out_size, void* d_ws, size_t ws_size,
                              hipStream_t stream) {
    const int*   positions = (const int*)d_in[0];
    const float* hidden    = (const float*)d_in[1];
    const float* Wqkv      = (const float*)d_in[2];
    const float* bqkv      = (const float*)d_in[3];
    const float* Wo        = (const float*)d_in[4];
    const float* bo        = (const float*)d_in[5];
    float*       out       = (float*)d_out;

    char* ws = (char*)d_ws;
    __bf16* hid_bf = (__bf16*)ws;                                    // 16 MB
    __bf16* qkv_bf = (__bf16*)(ws + (size_t)S_LEN * HID * 2);        // 48 MB
    __bf16* att_bf = (__bf16*)(ws + (size_t)S_LEN * HID * 2
                                  + (size_t)S_LEN * QKV_W * 2);      // 16 MB

    int n_hid = S_LEN * HID;
    cvt_f32_bf16_kernel<<<(n_hid + 255) / 256, 256, 0, stream>>>(hidden, hid_bf, n_hid);

    gemm_bias_kernel<true><<<dim3(QKV_W / GBN, S_LEN / GBM), 256, 0, stream>>>(
        hid_bf, Wqkv, bqkv, (void*)qkv_bf, S_LEN, QKV_W, HID);

    int n_rope = S_LEN * NHEAD * 64;
    rope_kernel<<<(n_rope + 255) / 256, 256, 0, stream>>>(qkv_bf, positions);

    attn_kernel<<<dim3(S_LEN / 64, NHEAD), 128, 0, stream>>>(qkv_bf, att_bf);

    gemm_bias_kernel<false><<<dim3(HID / GBN, S_LEN / GBM), 256, 0, stream>>>(
        att_bf, Wo, bo, (void*)out, S_LEN, HID, HID);
}
